// MultiHeadAttention_14207751815736
// MI455X (gfx1250) — compile-verified, matrix operand order checked by reading, not run on hardware
//
#include <hip/hip_runtime.h>
#include <hip/hip_bf16.h>
#include <math.h>

// ---------------------------------------------------------------------------
// MHA forward for gfx1250 (CDNA5): B=2, L=2048, D=1024, H=16, DK=64.
// Flash-attention structure: scores never touch HBM. All matmuls via
// v_wmma_f32_16x16x32_f16 (wave32, f16 A/B, f32 accumulate).
// K-tile staging in attention uses the Tensor Data Mover (tensor_load_to_lds,
// 6-arg clang-23 form); GEMM weight tiles are LDS-staged and shared by all
// 8 waves of a block to cut L2 traffic 8x.
// ---------------------------------------------------------------------------

typedef _Float16 half_t;
typedef __attribute__((ext_vector_type(16))) _Float16 v16h;
typedef __attribute__((ext_vector_type(8)))  float    v8f;
typedef __attribute__((ext_vector_type(4)))  unsigned int v4u;
typedef __attribute__((ext_vector_type(8)))  int      v8i;
typedef __attribute__((ext_vector_type(4)))  int      v4i;

#define BATCH 2
#define SEQ   2048
#define DIM   1024
#define HEADS 16
#define DKH   64

// ---- WMMA fragment helpers -------------------------------------------------
// A / B fragment (16-bit, 16x32 / 32x16): lane (m = lane&15) holds 16 halves,
// K index pattern: h<8 -> k = h + 8*g ; h>=8 -> k = h + 8 + 8*g  (g = lane>>4)
__device__ __forceinline__ int frag_k(int h, int g) {
  return (h < 8) ? (h + 8 * g) : (h + 8 + 8 * g);
}

__device__ __forceinline__ v16h load_frag_f32(const float* __restrict__ base,
                                              int ld, int lane) {
  const int g = (lane >> 4) & 1;
  const float* p = base + (size_t)(lane & 15) * ld;
  v16h a;
#pragma unroll
  for (int h = 0; h < 16; ++h) a[h] = (half_t)p[frag_k(h, g)];
  return a;
}

__device__ __forceinline__ v16h load_frag_f16(const half_t* __restrict__ base,
                                              int ld, int lane) {
  const int g = (lane >> 4) & 1;
  const half_t* p = base + (size_t)(lane & 15) * ld;
  v16h a;
#pragma unroll
  for (int h = 0; h < 16; ++h) a[h] = p[frag_k(h, g)];
  return a;
}

__device__ __forceinline__ v8f wmma_f16(v16h a, v16h b, v8f c) {
  return __builtin_amdgcn_wmma_f32_16x16x32_f16(false, a, false, b,
                                                (short)0, c, false, false);
}

// ---- TDM helper: async 2D f16 tile (trows x tcols) from global to LDS ------
// Tensor is row-major [nrows][ld] f16; tile starts at gptr.
// D# layout per CDNA5 ISA ch.8 (group0: count/lds/global/type, group1: dims).
// clang-23 builtin arity: (v4u g0, v8i g1, v4i g2, v4i g3, v8i extra, i32 cpol)
#if __has_builtin(__builtin_amdgcn_tensor_load_to_lds) && \
    __has_builtin(__builtin_amdgcn_s_wait_tensorcnt)
#define HAVE_TDM 1
__device__ __forceinline__ void tdm_load_tile_f16(const half_t* gptr,
                                                  half_t* ldsptr,
                                                  int tcols, int trows,
                                                  int nrows, int ld) {
  const unsigned long long ga = (unsigned long long)(uintptr_t)gptr;
  const unsigned lds = (unsigned)(uintptr_t)ldsptr;  // low 32 bits = LDS offset
  v4u g0;
  g0[0] = 1u;                                   // count=1, is_restore=0
  g0[1] = lds;                                  // lds_addr (bytes)
  g0[2] = (unsigned)ga;                         // global_addr[31:0]
  g0[3] = (unsigned)((ga >> 32) & 0x1FFFFFFu)   // global_addr[56:32]
          | (2u << 30);                         // type = 2 ("image")
  v8i g1;
  g1[0] = 0x10000;                              // wg_mask=0, data_size=1 (2B)
  g1[1] = (ld & 0xFFFF) << 16;                  // tensor_dim0[15:0] (bits 63:48)
  g1[2] = (nrows & 0xFFFF) << 16;               // dim0[31:16]=0 | tensor_dim1 lo
  g1[3] = (tcols & 0xFFFF) << 16;               // dim1 hi=0 | tile_dim0
  g1[4] = trows & 0xFFFF;                       // tile_dim1 | tile_dim2=0
  g1[5] = ld;                                   // tensor_dim0_stride[31:0]
  g1[6] = 0;                                    // stride hi | dim1_stride lo
  g1[7] = 0;
  v4i gz4 = {0, 0, 0, 0};                       // groups 2/3 unused (2D tile)
  v8i gz8 = {0, 0, 0, 0, 0, 0, 0, 0};
  __builtin_amdgcn_tensor_load_to_lds(g0, g1, gz4, gz4, gz8, 0);
}
#else
#define HAVE_TDM 0
#endif

// ---------------------------------------------------------------------------
// Kernel 1: fused Q/K/V projection.  Y = X @ W^T + b  (M=4096, N=1024, K=1024)
// Block tile 128 rows x 64 cols (8 waves x 16 rows). Per k-step the 64x32 W
// tile is converted to f16 in LDS once and shared by all waves.
// Output f16, head-split layout [B, H, L, DK].
// ---------------------------------------------------------------------------
__global__ __launch_bounds__(256) void qkv_proj_kernel(
    const float* __restrict__ q, const float* __restrict__ k,
    const float* __restrict__ v,
    const float* __restrict__ Wq, const float* __restrict__ Wk,
    const float* __restrict__ Wv,
    const float* __restrict__ bq, const float* __restrict__ bk,
    const float* __restrict__ bv,
    half_t* __restrict__ Qh, half_t* __restrict__ Kh, half_t* __restrict__ Vh) {
  __shared__ half_t Wl[64 * 32];            // W tile: [n][k] f16

  const int lane = threadIdx.x & 31;
  const int wid  = threadIdx.x >> 5;

  const int mi = blockIdx.x / 512;          // 0..2 : which of Q/K/V
  const int t  = blockIdx.x % 512;          // 32 row-tiles x 16 col-tiles
  const int row0 = (t >> 4) * 128;
  const int col0 = (t & 15) * 64;
  const int wrow = row0 + wid * 16;         // this wave's 16 rows

  const float* X    = (mi == 0) ? q  : (mi == 1) ? k  : v;
  const float* W    = (mi == 0) ? Wq : (mi == 1) ? Wk : Wv;
  const float* bias = (mi == 0) ? bq : (mi == 1) ? bk : bv;
  half_t*      out  = (mi == 0) ? Qh : (mi == 1) ? Kh : Vh;

  v8f acc[4] = {};
  for (int k0 = 0; k0 < DIM; k0 += 32) {
    // stage W[col0+n][k0+kk] -> Wl (256 threads x 8 elems)
    {
      const int n  = threadIdx.x >> 2;
      const int kk = (threadIdx.x & 3) * 8;
      const float* wp = W + (size_t)(col0 + n) * DIM + k0 + kk;
      half_t* dst = &Wl[n * 32 + kk];
#pragma unroll
      for (int i = 0; i < 8; ++i) dst[i] = (half_t)wp[i];
    }
    __syncthreads();

    v16h a = load_frag_f32(X + (size_t)wrow * DIM + k0, DIM, lane);
#pragma unroll
    for (int j = 0; j < 4; ++j) {
      v16h bf = load_frag_f16(&Wl[(j * 16) * 32], 32, lane);
      acc[j] = wmma_f16(a, bf, acc[j]);
    }
    __syncthreads();
  }

  const int g = lane >> 4, nl = lane & 15;
#pragma unroll
  for (int j = 0; j < 4; ++j) {
#pragma unroll
    for (int r = 0; r < 8; ++r) {
      const int n   = col0 + j * 16 + nl;     // output feature
      const int row = wrow + r + 8 * g;       // flat (b*L + l)
      const float y = acc[j][r] + bias[n];
      const int bb = row >> 11, l = row & (SEQ - 1);
      const int hh = n >> 6, dk = n & 63;
      out[(((size_t)(bb * HEADS + hh)) * SEQ + l) * DKH + dk] = (half_t)y;
    }
  }
}

// ---------------------------------------------------------------------------
// Kernel 2: causal flash attention per (b, h, 64-row q tile).
// 4 waves x 16-row q strips; KV streamed in 32-wide chunks through LDS.
// K tile staged by the Tensor Data Mover (wave 0) when available.
// ---------------------------------------------------------------------------
__global__ __launch_bounds__(128) void flash_attn_kernel(
    const half_t* __restrict__ Qh, const half_t* __restrict__ Kh,
    const half_t* __restrict__ Vh, half_t* __restrict__ Ah) {
  const int bh = blockIdx.x >> 5;          // b*H + h
  const int q0 = (blockIdx.x & 31) * 64;   // q tile base
  const int lane = threadIdx.x & 31;
  const int wid  = threadIdx.x >> 5;
  const int qbase = q0 + wid * 16;         // this wave's 16 q rows
  const int g = lane >> 4, nl = lane & 15;

  __shared__ half_t Kt[32 * DKH];          // K chunk,  row-major [kk][d]
  __shared__ half_t Vt[DKH * 32];          // V chunk, transposed [d][kk]
  __shared__ half_t Pt[4][16 * 32];        // per-wave P tile re-layout buffer

  // Q strip resident in registers: 16 x 64 -> two A fragments
  const half_t* Qb = Qh + ((size_t)bh * SEQ + qbase) * DKH;
  const v16h aq0 = load_frag_f16(Qb +  0, DKH, lane);
  const v16h aq1 = load_frag_f16(Qb + 32, DKH, lane);

  float mrow[8], ssum[8];
#pragma unroll
  for (int r = 0; r < 8; ++r) { mrow[r] = -INFINITY; ssum[r] = 0.f; }
  v8f oacc[4] = {};

  const float scale = 0.125f;              // 1/sqrt(64)
  const int kend = q0 + 64;                // causal: kv chunks with k0 <= q_max

  for (int kc = 0; kc < kend; kc += 32) {
    // ---- stage K chunk (TDM async DMA) and V chunk (manual transpose) ----
#if HAVE_TDM
    if (wid == 0) {
      tdm_load_tile_f16(Kh + ((size_t)bh * SEQ + kc) * DKH, Kt,
                        /*tcols=*/DKH, /*trows=*/32, /*nrows=*/SEQ, /*ld=*/DKH);
    }
#endif
    {
      const int kk = threadIdx.x >> 2;
      const int dp = (threadIdx.x & 3) * 16;
      const half_t* vs = Vh + ((size_t)bh * SEQ + kc + kk) * DKH + dp;
#if !HAVE_TDM
      const half_t* ks = Kh + ((size_t)bh * SEQ + kc + kk) * DKH + dp;
#endif
#pragma unroll
      for (int i = 0; i < 16; ++i) {
#if !HAVE_TDM
        Kt[kk * DKH + dp + i] = ks[i];
#endif
        Vt[(dp + i) * 32 + kk] = vs[i];
      }
    }
#if HAVE_TDM
    if (wid == 0) __builtin_amdgcn_s_wait_tensorcnt(0);
#endif
    __syncthreads();

    // ---- S = Q K^T : two 16x16 subtiles, K-dim 64 = 2 WMMAs each ----
    v8f s0 = {}, s1 = {};
    {
      v16h b;
      b = load_frag_f16(Kt + 0 * DKH +  0, DKH, lane); s0 = wmma_f16(aq0, b, s0);
      b = load_frag_f16(Kt + 0 * DKH + 32, DKH, lane); s0 = wmma_f16(aq1, b, s0);
      b = load_frag_f16(Kt + 16 * DKH +  0, DKH, lane); s1 = wmma_f16(aq0, b, s1);
      b = load_frag_f16(Kt + 16 * DKH + 32, DKH, lane); s1 = wmma_f16(aq1, b, s1);
    }

    // ---- online softmax (rows split per half-wave; reduce over 16 lanes) ----
    half_t* P = Pt[wid];
#pragma unroll
    for (int r = 0; r < 8; ++r) {
      const int row = qbase + r + 8 * g;
      const int c0 = kc + nl, c1 = kc + 16 + nl;
      float v0 = (c0 <= row) ? s0[r] * scale : -INFINITY;
      float v1 = (c1 <= row) ? s1[r] * scale : -INFINITY;

      float mx = fmaxf(v0, v1);
#pragma unroll
      for (int off = 1; off < 16; off <<= 1)
        mx = fmaxf(mx, __shfl_xor(mx, off, 16));
      const float mnew = fmaxf(mrow[r], mx);
      const float alpha = __expf(mrow[r] - mnew);
      mrow[r] = mnew;

      const float p0 = __expf(v0 - mnew);
      const float p1 = __expf(v1 - mnew);
      float rs = p0 + p1;
#pragma unroll
      for (int off = 1; off < 16; off <<= 1)
        rs += __shfl_xor(rs, off, 16);
      ssum[r] = ssum[r] * alpha + rs;
#pragma unroll
      for (int f = 0; f < 4; ++f) oacc[f][r] *= alpha;

      // C-layout -> LDS so P can be re-read in A-fragment layout
      const int m = r + 8 * g;
      P[m * 32 + nl]      = (half_t)p0;
      P[m * 32 + 16 + nl] = (half_t)p1;
    }
    __syncthreads();

    // ---- O += P V : A from LDS P tile, B from transposed V tile ----
    {
      const v16h ap = load_frag_f16(P, 32, lane);
#pragma unroll
      for (int f = 0; f < 4; ++f) {
        v16h bv = load_frag_f16(Vt + (f * 16) * 32, 32, lane);
        oacc[f] = wmma_f16(ap, bv, oacc[f]);
      }
    }
    __syncthreads();
  }

  // ---- normalize and store attn output as f16 [B, L, D] (heads merged) ----
  const int bb = bh >> 4, hh = bh & 15;
#pragma unroll
  for (int f = 0; f < 4; ++f) {
#pragma unroll
    for (int r = 0; r < 8; ++r) {
      const int row = qbase + r + 8 * g;            // l index
      const int d   = hh * DKH + f * 16 + nl;       // feature index
      const float o = oacc[f][r] / ssum[r];
      Ah[((size_t)bb * SEQ + row) * DIM + d] = (half_t)o;
    }
  }
}

// ---------------------------------------------------------------------------
// Kernel 3: output projection.  Y = A @ Wo^T + bo  -> f32 d_out
// Same LDS-staged weight-tile structure as kernel 1.
// ---------------------------------------------------------------------------
__global__ __launch_bounds__(256) void out_proj_kernel(
    const half_t* __restrict__ Ah, const float* __restrict__ Wo,
    const float* __restrict__ bo, float* __restrict__ Y) {
  __shared__ half_t Wl[64 * 32];

  const int lane = threadIdx.x & 31;
  const int wid  = threadIdx.x >> 5;
  const int row0 = (blockIdx.x >> 4) * 128;
  const int col0 = (blockIdx.x & 15) * 64;
  const int wrow = row0 + wid * 16;

  v8f acc[4] = {};
  for (int k0 = 0; k0 < DIM; k0 += 32) {
    {
      const int n  = threadIdx.x >> 2;
      const int kk = (threadIdx.x & 3) * 8;
      const float* wp = Wo + (size_t)(col0 + n) * DIM + k0 + kk;
      half_t* dst = &Wl[n * 32 + kk];
#pragma unroll
      for (int i = 0; i < 8; ++i) dst[i] = (half_t)wp[i];
    }
    __syncthreads();

    v16h a = load_frag_f16(Ah + (size_t)wrow * DIM + k0, DIM, lane);
#pragma unroll
    for (int j = 0; j < 4; ++j) {
      v16h bf = load_frag_f16(&Wl[(j * 16) * 32], 32, lane);
      acc[j] = wmma_f16(a, bf, acc[j]);
    }
    __syncthreads();
  }

  const int g = lane >> 4, nl = lane & 15;
#pragma unroll
  for (int j = 0; j < 4; ++j) {
#pragma unroll
    for (int r = 0; r < 8; ++r) {
      const int n   = col0 + j * 16 + nl;
      const int row = wrow + r + 8 * g;
      Y[(size_t)row * DIM + n] = acc[j][r] + bo[n];
    }
  }
}

// ---------------------------------------------------------------------------
extern "C" void kernel_launch(void* const* d_in, const int* in_sizes, int n_in,
                              void* d_out, int out_size, void* d_ws, size_t ws_size,
                              hipStream_t stream) {
  const float* q  = (const float*)d_in[0];
  const float* k  = (const float*)d_in[1];
  const float* v  = (const float*)d_in[2];
  // d_in[3] = mask: exactly causal triu(k=1); handled analytically in-kernel.
  const float* Wq = (const float*)d_in[4];
  const float* bq = (const float*)d_in[5];
  const float* Wk = (const float*)d_in[6];
  const float* bk = (const float*)d_in[7];
  const float* Wv = (const float*)d_in[8];
  const float* bv = (const float*)d_in[9];
  const float* Wo = (const float*)d_in[10];
  const float* bo = (const float*)d_in[11];
  float* out = (float*)d_out;

  // Workspace: 4 x 8 MB f16 buffers
  char* ws = (char*)d_ws;
  const size_t seg = (size_t)BATCH * HEADS * SEQ * DKH * sizeof(half_t); // 8 MB
  half_t* Qh = (half_t*)(ws + 0 * seg);
  half_t* Kh = (half_t*)(ws + 1 * seg);
  half_t* Vh = (half_t*)(ws + 2 * seg);
  half_t* Ah = (half_t*)(ws + 3 * seg);

  // 1) QKV projections: 3 matrices x 512 block-tiles (128x64), 8 waves/block
  qkv_proj_kernel<<<1536, 256, 0, stream>>>(q, k, v, Wq, Wk, Wv, bq, bk, bv,
                                            Qh, Kh, Vh);
  // 2) flash attention: (B*H) * (L/64) blocks, 4 waves each
  flash_attn_kernel<<<BATCH * HEADS * (SEQ / 64), 128, 0, stream>>>(Qh, Kh, Vh, Ah);
  // 3) output projection: 512 block-tiles, 8 waves/block
  out_proj_kernel<<<512, 256, 0, stream>>>(Ah, Wo, bo, out);
}
